// DomainAdapter_28939489640779
// MI455X (gfx1250) — matrix operand order repaired and checked by reading, not compiled, not run on hardware
//
#include <hip/hip_runtime.h>
#include <hip/hip_bf16.h>
#include <math.h>

// ---------------------------------------------------------------------------
// DomainAdapter fused kernel for gfx1250 (MI455X), bf16 WMMA + f32 accumulate.
// Weights pre-converted to bf16 in workspace; each wave owns BOTH M-tiles so
// every B-fragment global load feeds 2 WMMAs (no duplicate weight traffic).
// ---------------------------------------------------------------------------

typedef __bf16 bf16_t;
typedef __attribute__((ext_vector_type(16))) __bf16 v16bf;
typedef __attribute__((ext_vector_type(8)))  __bf16 v8bf;
typedef __attribute__((ext_vector_type(4)))  __bf16 v4bf;
typedef __attribute__((ext_vector_type(8)))  float  v8f;

#define D_DIM 1024
#define A_DIM 256
#define E_DIM 16
#define ROWS  32                 // seq rows per block (2 M-tiles)
#define SX_STRIDE 1032           // 1024 + 8 bf16 pad (row step 516 dwords, %64 = 4)
#define SH_STRIDE 264            // 256  + 8 bf16 pad (row step 132 dwords, %64 = 4)
#define LN_EPS 1e-5f

// bf16 element counts of the weight cache in d_ws
#define DW_ELEMS (A_DIM * D_DIM)            // 262144
#define UW_ELEMS (D_DIM * A_DIM)            // 262144
#define EW_ELEMS (E_DIM * A_DIM * A_DIM)    // 1048576
#define WS_NEED  ((size_t)(DW_ELEMS + UW_ELEMS + EW_ELEMS) * sizeof(bf16_t))

__device__ __forceinline__ float gelu_exact(float x) {
  // torch nn.GELU default: 0.5*x*(1+erf(x/sqrt(2)))
  return 0.5f * x * (1.0f + erff(x * 0.70710678118654752f));
}

// A-fragment: 16x32 bf16 tile (M x K) from LDS, contiguous-K layout (same k order as frag_b):
// lanes 0-15: M=lane, K=0..15; lanes 16-31: M=lane-16, K=16..31.
__device__ __forceinline__ v16bf frag_a_lds(const bf16_t* p0, int stride, int kBase, int lane) {
  const int m = lane & 15;
  const int khalf = (lane >> 4) << 4;          // 0 or 16
  const bf16_t* p = p0 + m * stride + kBase + khalf;
  v8bf lo = *reinterpret_cast<const v8bf*>(p);       // K offsets +0..7
  v8bf hi = *reinterpret_cast<const v8bf*>(p + 8);   // K offsets +8..15
  v16bf a;
#pragma unroll
  for (int i = 0; i < 8; ++i) { a[i] = lo[i]; a[8 + i] = hi[i]; }
  return a;
}

// B-fragment row loaders: 16 contiguous K values of one weight row.
__device__ __forceinline__ v16bf frag_b_row(const float* __restrict__ p) {
  const float4* q = reinterpret_cast<const float4*>(p);
  v16bf bv;
#pragma unroll
  for (int i = 0; i < 4; ++i) {
    float4 f = q[i];
    bv[4 * i + 0] = (bf16_t)f.x;
    bv[4 * i + 1] = (bf16_t)f.y;
    bv[4 * i + 2] = (bf16_t)f.z;
    bv[4 * i + 3] = (bf16_t)f.w;
  }
  return bv;
}
__device__ __forceinline__ v16bf frag_b_row(const bf16_t* __restrict__ p) {
  v8bf lo = *reinterpret_cast<const v8bf*>(p);
  v8bf hi = *reinterpret_cast<const v8bf*>(p + 8);
  v16bf bv;
#pragma unroll
  for (int i = 0; i < 8; ++i) { bv[i] = lo[i]; bv[8 + i] = hi[i]; }
  return bv;
}

// B-fragment: 32x16 tile (K x N): B[k][n] = W[nBase+n][kBase+k], W row-major.
// lanes 0-15: N=lane, K=0..15; lanes 16-31: N=lane-16, K=16..31.
template <typename WT>
__device__ __forceinline__ v16bf frag_b(const WT* __restrict__ W, int ldw,
                                        int nBase, int kBase, int lane) {
  const int n  = lane & 15;
  const int kb = kBase + ((lane >> 4) << 4);
  return frag_b_row(W + (size_t)(nBase + n) * ldw + kb);
}

__device__ __forceinline__ v8f wmma_bf16f32(v16bf a, v16bf b, v8f c) {
  return __builtin_amdgcn_wmma_f32_16x16x32_bf16(
      false, a, false, b, (short)0, c, false, false);
}

// ---------------------------------------------------------------------------
// f32 -> bf16 weight conversion (workspace cache); one float4 -> one v4bf.
// ---------------------------------------------------------------------------
__global__ void __launch_bounds__(256)
cvt_f32_to_bf16(const float* __restrict__ src, bf16_t* __restrict__ dst, int n4) {
  int i = blockIdx.x * 256 + threadIdx.x;
  if (i < n4) {
    float4 f = reinterpret_cast<const float4*>(src)[i];
    v4bf o;
    o[0] = (bf16_t)f.x; o[1] = (bf16_t)f.y; o[2] = (bf16_t)f.z; o[3] = (bf16_t)f.w;
    *reinterpret_cast<v4bf*>(dst + (size_t)i * 4) = o;
  }
}

// ---------------------------------------------------------------------------
// Kernel 1: down-proj + GELU + expert GEMM + emb + up-proj + bias + residual.
// One block = 32 seq rows of one batch (2 M-tiles), 8 waves.
//   GEMM1/2: wave w -> cols 32w..32w+31 (2 N-tiles) x both M-tiles, 4 accs
//   GEMM3:   wave w -> cols 128w..128w+127, 2 passes of (4 N x 2 M), 8 accs
// ---------------------------------------------------------------------------
template <typename WT>
__global__ void __launch_bounds__(256)
adapter_fused(const float* __restrict__ X,
              const int*   __restrict__ domain_id,
              const WT*    __restrict__ downW,  // [256,1024]
              const float* __restrict__ downB,  // [256]
              const WT*    __restrict__ upW,    // [1024,256]
              const float* __restrict__ upB,    // [1024]
              const WT*    __restrict__ expW,   // [16,256,256]
              const float* __restrict__ expB,   // [16,256]
              const float* __restrict__ demb,   // [16,256]
              float* __restrict__ out) {
  extern __shared__ char smem_raw[];
  bf16_t* sX = reinterpret_cast<bf16_t*>(smem_raw);                    // [32][1032]
  bf16_t* sH = reinterpret_cast<bf16_t*>(smem_raw) + ROWS * SX_STRIDE; // [32][264]

  const int tid  = threadIdx.x;
  const int lane = tid & 31;
  const int w    = tid >> 5;
  const int chunk = blockIdx.x & 31;          // S/ROWS = 32
  const int b     = blockIdx.x >> 5;
  const int s0    = chunk * ROWS;
  const float* Xblk = X + ((size_t)b * 1024 + s0) * (size_t)D_DIM;

  // ---- stage X tile -> bf16 LDS (coalesced float4) ----
#pragma unroll 8
  for (int j = 0; j < 32; ++j) {
    int idx = j * 256 + tid;                  // 8192 float4s
    int r  = idx >> 8;                        // 256 float4 per row
    int c4 = idx & 255;
    float4 f = reinterpret_cast<const float4*>(Xblk + (size_t)r * D_DIM)[c4];
    bf16_t* p = &sX[r * SX_STRIDE + c4 * 4];
    p[0] = (bf16_t)f.x; p[1] = (bf16_t)f.y; p[2] = (bf16_t)f.z; p[3] = (bf16_t)f.w;
  }
  __syncthreads();

  const v8f vzero = {};
  const int n   = lane & 15;
  const int mhi = (lane >> 4) << 3;

  // ================= GEMM1: H = GELU(X @ downW^T + downB) =================
  {
    const int colBase = w * 32;               // wave's 2 N-tiles
    v8f acc[2][2];                            // [M-tile][N-tile]
#pragma unroll
    for (int mi = 0; mi < 2; ++mi)
#pragma unroll
      for (int t = 0; t < 2; ++t) acc[mi][t] = vzero;
    for (int kc = 0; kc < D_DIM; kc += 32) {
      v16bf a0 = frag_a_lds(sX, SX_STRIDE, kc, lane);
      v16bf a1 = frag_a_lds(sX + 16 * SX_STRIDE, SX_STRIDE, kc, lane);
#pragma unroll
      for (int t = 0; t < 2; ++t) {
        v16bf bb = frag_b(downW, D_DIM, colBase + t * 16, kc, lane);
        acc[0][t] = wmma_bf16f32(a0, bb, acc[0][t]);
        acc[1][t] = wmma_bf16f32(a1, bb, acc[1][t]);
      }
    }
#pragma unroll
    for (int t = 0; t < 2; ++t) {
      const int col = colBase + t * 16 + n;
      const float bias = downB[col];
#pragma unroll
      for (int mi = 0; mi < 2; ++mi)
#pragma unroll
        for (int r = 0; r < 8; ++r) {
          int m = mi * 16 + r + mhi;
          sH[m * SH_STRIDE + col] = (bf16_t)gelu_exact(acc[mi][t][r] + bias);
        }
    }
  }
  __syncthreads();

  // ============ GEMM2: expert: H2 = where(valid, H@We^T+eb, H) + emb =======
  const int id_raw = domain_id[b];
  const bool valid = (id_raw >= 0) && (id_raw < E_DIM);
  const int  idx   = id_raw < 0 ? 0 : (id_raw > E_DIM - 1 ? E_DIM - 1 : id_raw);
  const WT* We     = expW + (size_t)idx * A_DIM * A_DIM;

  {
    const int colBase = w * 32;
    v8f e[2][2];
#pragma unroll
    for (int mi = 0; mi < 2; ++mi)
#pragma unroll
      for (int t = 0; t < 2; ++t) e[mi][t] = vzero;
    for (int kc = 0; kc < A_DIM; kc += 32) {
      v16bf a0 = frag_a_lds(sH, SH_STRIDE, kc, lane);
      v16bf a1 = frag_a_lds(sH + 16 * SH_STRIDE, SH_STRIDE, kc, lane);
#pragma unroll
      for (int t = 0; t < 2; ++t) {
        v16bf bb = frag_b(We, A_DIM, colBase + t * 16, kc, lane);
        e[0][t] = wmma_bf16f32(a0, bb, e[0][t]);
        e[1][t] = wmma_bf16f32(a1, bb, e[1][t]);
      }
    }
    float h2[2][2][8];
#pragma unroll
    for (int t = 0; t < 2; ++t) {
      const int col = colBase + t * 16 + n;
      const float eb = expB[idx * A_DIM + col];
      const float de = demb[idx * A_DIM + col];
#pragma unroll
      for (int mi = 0; mi < 2; ++mi)
#pragma unroll
        for (int r = 0; r < 8; ++r) {
          int m = mi * 16 + r + mhi;
          float v = valid ? (e[mi][t][r] + eb) : (float)sH[m * SH_STRIDE + col];
          h2[mi][t][r] = v + de;
        }
    }
    __syncthreads();                          // everyone done reading H
#pragma unroll
    for (int t = 0; t < 2; ++t) {
      const int col = colBase + t * 16 + n;
#pragma unroll
      for (int mi = 0; mi < 2; ++mi)
#pragma unroll
        for (int r = 0; r < 8; ++r) {
          int m = mi * 16 + r + mhi;
          sH[m * SH_STRIDE + col] = (bf16_t)h2[mi][t][r];
        }
    }
  }
  __syncthreads();                            // H2 fully written

  // ======== GEMM3: Y = H2 @ upW^T + upB + residual (pre-LayerNorm) =========
  for (int pass = 0; pass < 2; ++pass) {
    const int colBase = w * 128 + pass * 64;  // 4 N-tiles this pass
    v8f o[2][4];
#pragma unroll
    for (int mi = 0; mi < 2; ++mi)
#pragma unroll
      for (int t = 0; t < 4; ++t) o[mi][t] = vzero;
    for (int kc = 0; kc < A_DIM; kc += 32) {
      v16bf a0 = frag_a_lds(sH, SH_STRIDE, kc, lane);
      v16bf a1 = frag_a_lds(sH + 16 * SH_STRIDE, SH_STRIDE, kc, lane);
#pragma unroll
      for (int t = 0; t < 4; ++t) {
        v16bf bb = frag_b(upW, A_DIM, colBase + t * 16, kc, lane);
        o[0][t] = wmma_bf16f32(a0, bb, o[0][t]);
        o[1][t] = wmma_bf16f32(a1, bb, o[1][t]);
      }
    }
#pragma unroll
    for (int t = 0; t < 4; ++t) {
      const int d = colBase + t * 16 + n;
      const float ub = upB[d];
#pragma unroll
      for (int mi = 0; mi < 2; ++mi)
#pragma unroll
        for (int r = 0; r < 8; ++r) {
          int m = mi * 16 + r + mhi;
          size_t off = ((size_t)b * 1024 + s0 + m) * (size_t)D_DIM + d;
          out[off] = o[mi][t][r] + ub + X[off];   // residual add
        }
    }
  }
}

// ---------------------------------------------------------------------------
// Kernel 2: in-place LayerNorm over last dim (1024). One block per row.
// ---------------------------------------------------------------------------
__global__ void __launch_bounds__(256)
layernorm_kernel(float* __restrict__ out,
                 const float* __restrict__ gamma,
                 const float* __restrict__ beta) {
  const int t = threadIdx.x;
  float* x = out + (size_t)blockIdx.x * D_DIM;

  float4 f = reinterpret_cast<const float4*>(x)[t];
  float s  = f.x + f.y + f.z + f.w;
  float s2 = f.x * f.x + f.y * f.y + f.z * f.z + f.w * f.w;

#pragma unroll
  for (int m = 16; m >= 1; m >>= 1) {
    s  += __shfl_xor(s,  m, 32);
    s2 += __shfl_xor(s2, m, 32);
  }
  __shared__ float rs[8], rs2[8];
  if ((t & 31) == 0) { rs[t >> 5] = s; rs2[t >> 5] = s2; }
  __syncthreads();
  float ts = 0.0f, ts2 = 0.0f;
#pragma unroll
  for (int i = 0; i < 8; ++i) { ts += rs[i]; ts2 += rs2[i]; }

  const float mu  = ts * (1.0f / (float)D_DIM);
  const float var = ts2 * (1.0f / (float)D_DIM) - mu * mu;
  const float rsd = rsqrtf(var + LN_EPS);

  float4 g  = reinterpret_cast<const float4*>(gamma)[t];
  float4 bt = reinterpret_cast<const float4*>(beta)[t];
  f.x = (f.x - mu) * rsd * g.x + bt.x;
  f.y = (f.y - mu) * rsd * g.y + bt.y;
  f.z = (f.z - mu) * rsd * g.z + bt.z;
  f.w = (f.w - mu) * rsd * g.w + bt.w;
  reinterpret_cast<float4*>(x)[t] = f;
}

// ---------------------------------------------------------------------------
extern "C" void kernel_launch(void* const* d_in, const int* in_sizes, int n_in,
                              void* d_out, int out_size, void* d_ws, size_t ws_size,
                              hipStream_t stream) {
  (void)in_sizes; (void)n_in; (void)out_size;
  const float* X      = (const float*)d_in[0];
  const int*   domid  = (const int*)  d_in[1];
  const float* downW  = (const float*)d_in[2];
  const float* downB  = (const float*)d_in[3];
  const float* upW    = (const float*)d_in[4];
  const float* upB    = (const float*)d_in[5];
  const float* expW   = (const float*)d_in[6];
  const float* expB   = (const float*)d_in[7];
  const float* demb   = (const float*)d_in[8];
  const float* gamma  = (const float*)d_in[9];
  const float* beta   = (const float*)d_in[10];
  float* out = (float*)d_out;

  const size_t lds_bytes = (size_t)(ROWS * SX_STRIDE + ROWS * SH_STRIDE) * sizeof(bf16_t);
  const int nblocks = 32 * (1024 / ROWS);     // 1024

  if (ws_size >= WS_NEED) {
    // cache all weights as bf16 in workspace (done every call: deterministic)
    bf16_t* wsb = (bf16_t*)d_ws;
    bf16_t* dwB = wsb;
    bf16_t* uwB = wsb + DW_ELEMS;
    bf16_t* ewB = wsb + DW_ELEMS + UW_ELEMS;
    cvt_f32_to_bf16<<<(DW_ELEMS / 4 + 255) / 256, 256, 0, stream>>>(downW, dwB, DW_ELEMS / 4);
    cvt_f32_to_bf16<<<(UW_ELEMS / 4 + 255) / 256, 256, 0, stream>>>(upW,   uwB, UW_ELEMS / 4);
    cvt_f32_to_bf16<<<(EW_ELEMS / 4 + 255) / 256, 256, 0, stream>>>(expW,  ewB, EW_ELEMS / 4);
    adapter_fused<bf16_t><<<nblocks, 256, lds_bytes, stream>>>(
        X, domid, dwB, downB, uwB, upB, ewB, expB, demb, out);
  } else {
    adapter_fused<float><<<nblocks, 256, lds_bytes, stream>>>(
        X, domid, downW, downB, upW, upB, expW, expB, demb, out);
  }
  layernorm_kernel<<<32 * 1024, 256, 0, stream>>>(out, gamma, beta);
}
